// MLA_7670811590815
// MI455X (gfx1250) — compile-verified, matrix operand order checked by reading, not run on hardware
//
#include <hip/hip_runtime.h>
#include <hip/hip_bf16.h>
#include <math.h>

// ---------------------------------------------------------------------------
// MLA (DeepSeek-style) forward, MI455X / gfx1250, wave32 + WMMA bf16.
// All matmuls in bf16 with f32 accumulate via v_wmma_f32_16x16x32_bf16.
// Softmax / RMS-norm / RoPE in f32 VALU. Causal mask applied analytically.
// GEMM: register-blocked 32x64 per wave (8 WMMAs per 6 fragment loads).
// ---------------------------------------------------------------------------

typedef __bf16 bf16_t;
typedef __attribute__((ext_vector_type(16))) __bf16 v16bf_t;
typedef __attribute__((ext_vector_type(8)))  __bf16 v8bf_t;
typedef __attribute__((ext_vector_type(8)))  float  v8f_t;

#define S_LEN   4096
#define DIM_    2048
#define NH      16
#define DQK     192
#define DNOPE   128
#define DROPE   64
#define DV      128
#define KLORA   512
#define HQ      (NH*DQK)          /* 3072 */
#define KVA_N   (KLORA+DROPE)     /* 576  */
#define KVUP_N  (NH*(DNOPE+DV))   /* 4096 */
#define ATT_N   (NH*DV)           /* 2048 */

#define V8F_ZERO {0.f,0.f,0.f,0.f,0.f,0.f,0.f,0.f}

// Load one 16x32 bf16 WMMA fragment (A-layout; B uses the same K-packed
// per-lane pattern from a pre-transposed N x K matrix).
// Lane l: l16 = l&15 selects the row (A) / column (B); hf = l>>4.
// Elements 0..7  <- K = k0 + 8*hf + 0..7      (one 16-byte load)
// Elements 8..15 <- K = k0 + 16 + 8*hf + 0..7 (one 16-byte load)
__device__ __forceinline__ v16bf_t frag_load(const bf16_t* base, int ld,
                                             int row0, int k0, int l16, int hf) {
  const bf16_t* p = base + (size_t)(row0 + l16) * (size_t)ld + (size_t)(k0 + 8 * hf);
  v8bf_t lo = *reinterpret_cast<const v8bf_t*>(p);
  v8bf_t hi = *reinterpret_cast<const v8bf_t*>(p + 16);
  return __builtin_shufflevector(lo, hi, 0,1,2,3,4,5,6,7,8,9,10,11,12,13,14,15);
}

// ---------------------------------------------------------------------------
// Register-blocked bf16 WMMA GEMM: C(MxN) = A(MxK bf16) * BT(NxK bf16).
// One wave computes a 32x64 block of C: 2 A-frags x 4 B-frags -> 8 WMMAs
// per 32-deep k-step. Requires M % 32 == 0, N % 64 == 0, K % 32 == 0.
// ---------------------------------------------------------------------------
template <typename OutT>
__launch_bounds__(256)
__global__ void wmma_gemm_kernel(const bf16_t* __restrict__ A,
                                 const bf16_t* __restrict__ BT,
                                 OutT* __restrict__ C, int M, int N, int K) {
  const int lane = threadIdx.x & 31;
  const int wave = threadIdx.x >> 5;
  const int l16  = lane & 15;
  const int hf   = lane >> 4;
  const int bn   = N >> 6;                               // 64-wide col blocks
  const long blk   = (long)blockIdx.x * (long)(blockDim.x >> 5) + wave;
  const long total = (long)(M >> 5) * (long)bn;
  if (blk >= total) return;
  const int row0 = (int)(blk / bn) << 5;
  const int col0 = (int)(blk % bn) << 6;

  v8f_t acc[2][4];
  #pragma unroll
  for (int mi = 0; mi < 2; ++mi)
    #pragma unroll
    for (int ni = 0; ni < 4; ++ni) acc[mi][ni] = (v8f_t)V8F_ZERO;

  for (int k0 = 0; k0 < K; k0 += 32) {
    v16bf_t a0 = frag_load(A, K, row0,      k0, l16, hf);
    v16bf_t a1 = frag_load(A, K, row0 + 16, k0, l16, hf);
    __builtin_prefetch(A + (size_t)(row0 + l16) * K + k0 + 64, 0, 1);
    #pragma unroll
    for (int ni = 0; ni < 4; ++ni) {
      v16bf_t b = frag_load(BT, K, col0 + ni * 16, k0, l16, hf);
      acc[0][ni] = __builtin_amdgcn_wmma_f32_16x16x32_bf16(false, a0, false, b,
                                                           (short)0, acc[0][ni],
                                                           false, false);
      acc[1][ni] = __builtin_amdgcn_wmma_f32_16x16x32_bf16(false, a1, false, b,
                                                           (short)0, acc[1][ni],
                                                           false, false);
    }
  }
  // C layout per 16x16 tile: lane n = l16; VGPR i -> row i + 8*hf.
  #pragma unroll
  for (int mi = 0; mi < 2; ++mi) {
    #pragma unroll
    for (int ni = 0; ni < 4; ++ni) {
      OutT* cp = C + (size_t)(row0 + mi * 16 + 8 * hf) * (size_t)N
                   + (size_t)(col0 + ni * 16 + l16);
      #pragma unroll
      for (int i = 0; i < 8; ++i) cp[(size_t)i * N] = static_cast<OutT>(acc[mi][ni][i]);
    }
  }
}

// ---------------------------------------------------------------------------
// Elementwise f32 -> bf16 convert
// ---------------------------------------------------------------------------
__global__ void cvt_f32_bf16_kernel(const float* __restrict__ in,
                                    bf16_t* __restrict__ out, long n) {
  long i = (long)blockIdx.x * blockDim.x + threadIdx.x;
  long stride = (long)gridDim.x * blockDim.x;
  for (; i < n; i += stride) out[i] = (bf16_t)in[i];
}

// Transpose + convert: in = KxN f32 -> out = NxK bf16
__global__ void transpose_cvt_kernel(const float* __restrict__ in,
                                     bf16_t* __restrict__ out, int K, int N) {
  long i = (long)blockIdx.x * blockDim.x + threadIdx.x;
  long total = (long)K * N;
  long stride = (long)gridDim.x * blockDim.x;
  for (; i < total; i += stride) {
    int k = (int)(i % K);
    int n = (int)(i / K);
    out[(size_t)n * K + k] = (bf16_t)in[(size_t)k * N + n];
  }
}

// ---------------------------------------------------------------------------
// RMS-norm over the first 512 cols of kv (f32), one block per row.
// ---------------------------------------------------------------------------
__global__ void rmsnorm_kernel(const float* __restrict__ kv,
                               const float* __restrict__ w,
                               bf16_t* __restrict__ out, float eps) {
  const int t = blockIdx.x;
  const float* row = kv + (size_t)t * KVA_N;
  __shared__ float red[8];
  float ss = 0.f;
  for (int j = threadIdx.x; j < KLORA; j += blockDim.x) {
    float v = row[j];
    ss += v * v;
  }
  #pragma unroll
  for (int o = 16; o >= 1; o >>= 1) ss += __shfl_xor(ss, o, 32);
  if ((threadIdx.x & 31) == 0) red[threadIdx.x >> 5] = ss;
  __syncthreads();
  float tot = 0.f;
  #pragma unroll
  for (int i = 0; i < 8; ++i) tot += red[i];
  const float sc = rsqrtf(tot / (float)KLORA + eps);
  bf16_t* orow = out + (size_t)t * KLORA;
  for (int j = threadIdx.x; j < KLORA; j += blockDim.x)
    orow[j] = (bf16_t)(row[j] * sc * w[j]);
}

// ---------------------------------------------------------------------------
// RoPE on q: q_full f32 [S][NH*192] -> qb bf16 [S][NH*192]
// (cols 0..127 copied, cols 128..191 rotated with cos/sin[s][32]).
// One thread per (s, h, j) with j in [0,96): j<64 copies a pair, else rotates.
// ---------------------------------------------------------------------------
__global__ void rope_q_kernel(const float* __restrict__ qf,
                              const float* __restrict__ cosb,
                              const float* __restrict__ sinb,
                              bf16_t* __restrict__ qb) {
  long idx = (long)blockIdx.x * blockDim.x + threadIdx.x;
  const long total = (long)S_LEN * NH * 96;
  if (idx >= total) return;
  const int j = (int)(idx % 96);
  long r = idx / 96;
  const int h = (int)(r % NH);
  const int s = (int)(r / NH);
  const float* in = qf + (size_t)s * HQ + (size_t)h * DQK;
  bf16_t* out = qb + (size_t)s * HQ + (size_t)h * DQK;
  if (j < 64) {
    out[2 * j]     = (bf16_t)in[2 * j];
    out[2 * j + 1] = (bf16_t)in[2 * j + 1];
  } else {
    const int p = j - 64;                       // 0..31
    const float xr = in[DNOPE + 2 * p];
    const float xi = in[DNOPE + 2 * p + 1];
    const float c  = cosb[(size_t)s * 32 + p];
    const float sn = sinb[(size_t)s * 32 + p];
    out[DNOPE + 2 * p]     = (bf16_t)(xr * c - xi * sn);
    out[DNOPE + 2 * p + 1] = (bf16_t)(xr * sn + xi * c);
  }
}

// ---------------------------------------------------------------------------
// Build K per head: kb bf16 [NH][S][192] = [ k_nope(kv_up) | rope(k_pe) ].
// k_nope from kvup bf16 [t][h*256 + 0..127]; k_pe from kv f32 [t][512..575].
// ---------------------------------------------------------------------------
__global__ void build_k_kernel(const bf16_t* __restrict__ kvup,
                               const float* __restrict__ kvf,
                               const float* __restrict__ cosb,
                               const float* __restrict__ sinb,
                               bf16_t* __restrict__ kb) {
  long idx = (long)blockIdx.x * blockDim.x + threadIdx.x;
  const long total = (long)S_LEN * NH * 96;
  if (idx >= total) return;
  const int j = (int)(idx % 96);
  long r = idx / 96;
  const int h = (int)(r % NH);
  const int t = (int)(r / NH);
  bf16_t* out = kb + ((size_t)h * S_LEN + t) * DQK;
  if (j < 64) {
    const bf16_t* in = kvup + (size_t)t * KVUP_N + (size_t)h * (DNOPE + DV);
    out[2 * j]     = in[2 * j];
    out[2 * j + 1] = in[2 * j + 1];
  } else {
    const int p = j - 64;
    const float* pe = kvf + (size_t)t * KVA_N + KLORA;
    const float xr = pe[2 * p];
    const float xi = pe[2 * p + 1];
    const float c  = cosb[(size_t)t * 32 + p];
    const float sn = sinb[(size_t)t * 32 + p];
    out[DNOPE + 2 * p]     = (bf16_t)(xr * c - xi * sn);
    out[DNOPE + 2 * p + 1] = (bf16_t)(xr * sn + xi * c);
  }
}

// Build V^T per head: vT bf16 [NH][DV][S] from kvup bf16 [t][h*256 + 128 + d]
__global__ void build_vT_kernel(const bf16_t* __restrict__ kvup,
                                bf16_t* __restrict__ vT) {
  long idx = (long)blockIdx.x * blockDim.x + threadIdx.x;
  const long total = (long)NH * DV * S_LEN;
  if (idx >= total) return;
  const int t = (int)(idx % S_LEN);
  long r = idx / S_LEN;
  const int d = (int)(r % DV);
  const int h = (int)(r / DV);
  vT[idx] = kvup[(size_t)t * KVUP_N + (size_t)h * (DNOPE + DV) + DNOPE + d];
}

// ---------------------------------------------------------------------------
// Flash attention, causal. One wave owns one (head, 16-query) tile, streams
// keys 32 at a time: 12 WMMAs for scores, online softmax, probs through LDS
// (C-layout -> A-layout transpose), 8 WMMAs for P*V against V^T.
// ---------------------------------------------------------------------------
__launch_bounds__(128)
__global__ void mla_flash_kernel(const bf16_t* __restrict__ Q,   // [S][HQ]
                                 const bf16_t* __restrict__ Kc,  // [NH][S][DQK]
                                 const bf16_t* __restrict__ VT,  // [NH][DV][S]
                                 bf16_t* __restrict__ O,         // [S][ATT_N]
                                 float scale) {
  __shared__ __align__(16) bf16_t plds[4][16 * 32];
  const int lane = threadIdx.x & 31;
  const int wave = threadIdx.x >> 5;
  const int l16  = lane & 15;
  const int hf   = lane >> 4;
  const int h    = blockIdx.y;
  const int q0   = (blockIdx.x * 4 + wave) << 4;

  const bf16_t* Qh = Q  + (size_t)h * DQK;
  const bf16_t* Kh = Kc + (size_t)h * S_LEN * DQK;
  const bf16_t* Vh = VT + (size_t)h * DV * S_LEN;

  v16bf_t qf[6];
  #pragma unroll
  for (int d = 0; d < 6; ++d) qf[d] = frag_load(Qh, HQ, q0, d * 32, l16, hf);

  v8f_t acc[8];
  #pragma unroll
  for (int j = 0; j < 8; ++j) acc[j] = (v8f_t)V8F_ZERO;
  float mrow[8], lrow[8];
  #pragma unroll
  for (int i = 0; i < 8; ++i) { mrow[i] = -1e30f; lrow[i] = 0.f; }

  const int kend = q0 + 16;  // causal: keys t <= q0+15
  for (int kb = 0; kb < kend; kb += 32) {
    v8f_t s0 = V8F_ZERO, s1 = V8F_ZERO;
    #pragma unroll
    for (int d = 0; d < 6; ++d) {
      v16bf_t kf = frag_load(Kh, DQK, kb, d * 32, l16, hf);
      s0 = __builtin_amdgcn_wmma_f32_16x16x32_bf16(false, qf[d], false, kf,
                                                   (short)0, s0, false, false);
    }
    #pragma unroll
    for (int d = 0; d < 6; ++d) {
      v16bf_t kf = frag_load(Kh, DQK, kb + 16, d * 32, l16, hf);
      s1 = __builtin_amdgcn_wmma_f32_16x16x32_bf16(false, qf[d], false, kf,
                                                   (short)0, s1, false, false);
    }
    // online softmax over 32 columns; row = q0 + i + 8*hf, col lane = l16
    float p0[8], p1[8], corr[8];
    #pragma unroll
    for (int i = 0; i < 8; ++i) {
      const int row = q0 + i + 8 * hf;
      float v0 = s0[i] * scale + ((kb + l16)      > row ? -1e30f : 0.f);
      float v1 = s1[i] * scale + ((kb + 16 + l16) > row ? -1e30f : 0.f);
      float mx = fmaxf(v0, v1);
      #pragma unroll
      for (int o = 1; o < 16; o <<= 1) mx = fmaxf(mx, __shfl_xor(mx, o, 32));
      const float nm = fmaxf(mrow[i], mx);
      const float c  = __expf(mrow[i] - nm);
      const float e0 = __expf(v0 - nm);
      const float e1 = __expf(v1 - nm);
      float rs = e0 + e1;
      #pragma unroll
      for (int o = 1; o < 16; o <<= 1) rs += __shfl_xor(rs, o, 32);
      lrow[i] = lrow[i] * c + rs;
      mrow[i] = nm;
      corr[i] = c; p0[i] = e0; p1[i] = e1;
    }
    #pragma unroll
    for (int j = 0; j < 8; ++j) {
      #pragma unroll
      for (int i = 0; i < 8; ++i) acc[j][i] *= corr[i];
    }
    // probs C-layout -> A-layout via per-wave LDS tile (16x32 bf16)
    bf16_t* L = plds[wave];
    #pragma unroll
    for (int i = 0; i < 8; ++i) {
      L[(i + 8 * hf) * 32 + l16]      = (bf16_t)p0[i];
      L[(i + 8 * hf) * 32 + 16 + l16] = (bf16_t)p1[i];
    }
    asm volatile("s_wait_dscnt 0x0" ::: "memory");
    v16bf_t pf = frag_load(L, 32, 0, 0, l16, hf);
    #pragma unroll
    for (int j = 0; j < 8; ++j) {
      v16bf_t vf = frag_load(Vh, S_LEN, j * 16, kb, l16, hf);
      acc[j] = __builtin_amdgcn_wmma_f32_16x16x32_bf16(false, pf, false, vf,
                                                       (short)0, acc[j], false, false);
    }
  }
  #pragma unroll
  for (int j = 0; j < 8; ++j) {
    bf16_t* op = O + (size_t)(q0 + 8 * hf) * ATT_N + (size_t)h * DV + j * 16 + l16;
    #pragma unroll
    for (int i = 0; i < 8; ++i)
      op[(size_t)i * ATT_N] = (bf16_t)(acc[j][i] / lrow[i]);
  }
}

// ---------------------------------------------------------------------------
// Launcher
// ---------------------------------------------------------------------------
extern "C" void kernel_launch(void* const* d_in, const int* in_sizes, int n_in,
                              void* d_out, int out_size, void* d_ws, size_t ws_size,
                              hipStream_t stream) {
  (void)in_sizes; (void)n_in; (void)out_size; (void)ws_size;
  const float* x      = (const float*)d_in[0];
  const float* fcos   = (const float*)d_in[1];
  const float* fsin   = (const float*)d_in[2];
  /* d_in[3] mask: causal mask applied analytically in-kernel */
  const float* wq     = (const float*)d_in[4];
  const float* wkv_a  = (const float*)d_in[5];
  const float* kvnw   = (const float*)d_in[6];
  const float* wkv_b  = (const float*)d_in[7];
  const float* wo     = (const float*)d_in[8];
  float* out = (float*)d_out;

  char* ws = (char*)d_ws;
  size_t off = 0;
  auto alloc = [&](size_t bytes) -> void* {
    void* p = ws + off;
    off = (off + bytes + 255) & ~(size_t)255;
    return p;
  };
  bf16_t* xbf    = (bf16_t*)alloc((size_t)S_LEN * DIM_ * 2);
  bf16_t* wqT    = (bf16_t*)alloc((size_t)HQ * DIM_ * 2);
  bf16_t* wkvaT  = (bf16_t*)alloc((size_t)KVA_N * DIM_ * 2);
  bf16_t* wkvbT  = (bf16_t*)alloc((size_t)KVUP_N * KLORA * 2);
  bf16_t* woT    = (bf16_t*)alloc((size_t)DIM_ * ATT_N * 2);
  float*  qfull  = (float*)alloc((size_t)S_LEN * HQ * 4);
  float*  kvf    = (float*)alloc((size_t)S_LEN * KVA_N * 4);
  bf16_t* cnorm  = (bf16_t*)alloc((size_t)S_LEN * KLORA * 2);
  bf16_t* kvup   = (bf16_t*)alloc((size_t)S_LEN * KVUP_N * 2);
  bf16_t* qb     = (bf16_t*)alloc((size_t)S_LEN * HQ * 2);
  bf16_t* kb     = (bf16_t*)alloc((size_t)NH * S_LEN * DQK * 2);
  bf16_t* vT     = (bf16_t*)alloc((size_t)NH * DV * S_LEN * 2);
  bf16_t* attn   = (bf16_t*)alloc((size_t)S_LEN * ATT_N * 2);

  const float mscale = 0.1f * logf(40.0f) + 1.0f;
  const float softmax_scale = powf((float)DQK, -0.5f) * mscale * mscale;

  // 1) convert / transpose-convert to bf16
  {
    long n = (long)S_LEN * DIM_;
    cvt_f32_bf16_kernel<<<(int)((n + 255) / 256), 256, 0, stream>>>(x, xbf, n);
    transpose_cvt_kernel<<<4096, 256, 0, stream>>>(wq,    wqT,   DIM_,  HQ);
    transpose_cvt_kernel<<<4096, 256, 0, stream>>>(wkv_a, wkvaT, DIM_,  KVA_N);
    transpose_cvt_kernel<<<4096, 256, 0, stream>>>(wkv_b, wkvbT, KLORA, KVUP_N);
    transpose_cvt_kernel<<<4096, 256, 0, stream>>>(wo,    woT,   ATT_N, DIM_);
  }
  auto gemm_blocks = [](int M, int N) {
    long blks = (long)(M / 32) * (N / 64);   // 32x64 per wave, 8 waves/block
    return (int)((blks + 7) / 8);
  };
  // 2) q_full = x @ wq  (f32 epilogue for RoPE)
  wmma_gemm_kernel<float><<<gemm_blocks(S_LEN, HQ), 256, 0, stream>>>(
      xbf, wqT, qfull, S_LEN, HQ, DIM_);
  // 3) kv = x @ wkv_a  (f32 epilogue for RMS-norm / RoPE)
  wmma_gemm_kernel<float><<<gemm_blocks(S_LEN, KVA_N), 256, 0, stream>>>(
      xbf, wkvaT, kvf, S_LEN, KVA_N, DIM_);
  // 4) RMS-norm of kv_lat
  rmsnorm_kernel<<<S_LEN, 256, 0, stream>>>(kvf, kvnw, cnorm, 1e-6f);
  // 5) kv_up = norm(kv_lat) @ wkv_b
  wmma_gemm_kernel<bf16_t><<<gemm_blocks(S_LEN, KVUP_N), 256, 0, stream>>>(
      cnorm, wkvbT, kvup, S_LEN, KVUP_N, KLORA);
  // 6) RoPE(q), build K per head, build V^T
  {
    long n = (long)S_LEN * NH * 96;
    rope_q_kernel<<<(int)((n + 255) / 256), 256, 0, stream>>>(qfull, fcos, fsin, qb);
    build_k_kernel<<<(int)((n + 255) / 256), 256, 0, stream>>>(kvup, kvf, fcos, fsin, kb);
    long nv = (long)NH * DV * S_LEN;
    build_vT_kernel<<<(int)((nv + 255) / 256), 256, 0, stream>>>(kvup, vT);
  }
  // 7) flash attention (causal), 4 waves/block, grid = (S/64, NH)
  mla_flash_kernel<<<dim3(S_LEN / 64, NH), 128, 0, stream>>>(
      qb, kb, vT, attn, softmax_scale);
  // 8) out = attn @ wo  (f32 epilogue to d_out)
  wmma_gemm_kernel<float><<<gemm_blocks(S_LEN, DIM_), 256, 0, stream>>>(
      attn, woT, out, S_LEN, DIM_, ATT_N);
}
